// SelfAttention_40561671143866
// MI455X (gfx1250) — matrix operand order repaired and checked by reading, not compiled, not run on hardware
//
#include <hip/hip_runtime.h>
#include <hip/hip_bf16.h>
#include <stdint.h>

// ---------------------------------------------------------------------------
// Self-attention for MI455X (gfx1250): bf16 WMMA GEMMs, double-buffered LDS
// fed by gfx1250 async global->LDS copies, + query-axis softmax.
//   x:[4,2048,1024] f32, Wq/Wk/Wv:[1024,1024] f32 -> out:[4,2048,1024] f32
// ---------------------------------------------------------------------------

typedef __attribute__((ext_vector_type(16))) __bf16 v16bf;
typedef __attribute__((ext_vector_type(8)))  __bf16 v8bf;
typedef __attribute__((ext_vector_type(8)))  float  v8f;
typedef int vi4 __attribute__((vector_size(16)));   // matches builtin param type

#define BM  128
#define BN  128
#define BKT 64          // K-slab per pipeline stage (2 WMMA K-steps)

#if __has_builtin(__builtin_amdgcn_global_load_async_to_lds_b128)
#define ASYNC_LDS 1
#else
#define ASYNC_LDS 0
#endif

#define AS1 __attribute__((address_space(1)))
#define AS3 __attribute__((address_space(3)))

static __device__ inline void async_stage_wait() {
#if ASYNC_LDS
#if __has_builtin(__builtin_amdgcn_s_wait_asynccnt)
    __builtin_amdgcn_s_wait_asynccnt(0);
#else
    asm volatile("s_wait_asynccnt 0x0" ::: "memory");
#endif
#endif
}

static __device__ inline void copy16B_to_lds(const __bf16* g, __bf16* l) {
#if ASYNC_LDS
    __builtin_amdgcn_global_load_async_to_lds_b128(
        (AS1 vi4*)(uintptr_t)g,
        (AS3 vi4*)(unsigned)(uintptr_t)l, 0, 0);
#else
    *(v8bf*)l = *(const v8bf*)g;
#endif
}

static __device__ inline v16bf cat8(v8bf lo, v8bf hi) {
    v16bf r;
#pragma unroll
    for (int i = 0; i < 8; ++i) { r[i] = lo[i]; r[i + 8] = hi[i]; }
    return r;
}

// ------------------------------ converters ---------------------------------

__global__ __launch_bounds__(256)
void cvt_f32_bf16(const float* __restrict__ src, __bf16* __restrict__ dst, int n) {
    int i = (blockIdx.x * 256 + threadIdx.x) * 4;
    if (i < n) {
        float4 v = *(const float4*)(src + i);
        dst[i + 0] = (__bf16)v.x;
        dst[i + 1] = (__bf16)v.y;
        dst[i + 2] = (__bf16)v.z;
        dst[i + 3] = (__bf16)v.w;
    }
}

// dst[c][r] = (bf16) src[r][c]
__global__ __launch_bounds__(256)
void cvt_f32_bf16_t(const float* __restrict__ src, __bf16* __restrict__ dst,
                    int rows, int cols) {
    int idx = blockIdx.x * 256 + threadIdx.x;
    if (idx < rows * cols) {
        int r = idx / cols;
        int c = idx - r * cols;
        dst[(long)c * rows + r] = (__bf16)src[idx];
    }
}

// ------------------------------- NT GEMM -----------------------------------
// C[m][n] = sum_k A[m][k] * B[n][k]   (A: MxK row-major, B: NxK row-major)
// cmode: 0 = bf16 row-major (ldc=N, + bz*strideC)
//        1 = bf16 transposed-batched: b=m>>11, out[b][n][m&2047] (2048 tokens)
//        2 = f32 row-major (ldc=N, + bz*strideC)

__global__ __launch_bounds__(256)
void gemm_nt_wmma(const __bf16* __restrict__ A, const __bf16* __restrict__ B,
                  void* __restrict__ C, int M, int N, int K,
                  long strideA, long strideB, long strideC, int cmode) {
    const int bz = blockIdx.z;
    A += (long)bz * strideA;
    B += (long)bz * strideB;

    __shared__ __bf16 Asl[2][BM * BKT];   // 2 x 16 KB
    __shared__ __bf16 Bsl[2][BN * BKT];   // 2 x 16 KB

    const int tid  = threadIdx.x;
    const int lane = tid & 31;
    const int wave = tid >> 5;      // 0..7
    const int wm   = wave >> 2;     // 0..1  -> 64-row slab
    const int wn   = wave & 3;      // 0..3  -> 32-col slab

    const int m0 = blockIdx.y * BM;
    const int n0 = blockIdx.x * BN;

    // staging map: each thread copies 64B (4 x b128) of A and of B per stage
    const int grow = tid >> 1;            // 0..127
    const int gcol = (tid & 1) << 5;      // 0 or 32 (elements)

    const int rsel = lane & 15;
    const int ksel = lane >> 4;           // 0/1

    v8f acc[4][2];
#pragma unroll
    for (int mi = 0; mi < 4; ++mi)
#pragma unroll
        for (int ni = 0; ni < 2; ++ni) acc[mi][ni] = (v8f){0,0,0,0,0,0,0,0};

    auto stage = [&](int buf, int k0) {
        const __bf16* ga = A + (long)(m0 + grow) * K + k0 + gcol;
        const __bf16* gb = B + (long)(n0 + grow) * K + k0 + gcol;
        __bf16* la = &Asl[buf][grow * BKT + gcol];
        __bf16* lb = &Bsl[buf][grow * BKT + gcol];
#pragma unroll
        for (int c = 0; c < 4; ++c) {
            copy16B_to_lds(ga + 8 * c, la + 8 * c);
            copy16B_to_lds(gb + 8 * c, lb + 8 * c);
        }
    };

    const int nIter = K / BKT;

    // prologue: fill buffer 0
    stage(0, 0);
    async_stage_wait();
    __syncthreads();

    for (int it = 0; it < nIter; ++it) {
        const int cur = it & 1;

        // fire next K-slab into the other buffer (async: overlaps with WMMA)
        if (it + 1 < nIter) {
            stage(cur ^ 1, (it + 1) * BKT);
            if (it + 2 < nIter) {   // warm L2 for the slab after that
                __builtin_prefetch(A + (long)(m0 + grow) * K + (it + 2) * BKT + gcol, 0, 3);
                __builtin_prefetch(B + (long)(n0 + grow) * K + (it + 2) * BKT + gcol, 0, 3);
            }
        }

        const __bf16* abuf = Asl[cur];
        const __bf16* bbuf = Bsl[cur];

#pragma unroll
        for (int kk = 0; kk < BKT; kk += 32) {
            // A fragments: 16x32 bf16, ISA layout
            //   lanes 0-15 : row=lane,    elems0-7=K0-7,  elems8-15=K16-23
            //   lanes16-31 : row=lane-16, elems0-7=K8-15, elems8-15=K24-31
            v16bf af[4];
#pragma unroll
            for (int mi = 0; mi < 4; ++mi) {
                int base = (wm * 64 + mi * 16 + rsel) * BKT + kk + ksel * 8;
                v8bf lo = *(const v8bf*)(&abuf[base]);
                v8bf hi = *(const v8bf*)(&abuf[base + 16]);
                af[mi]  = cat8(lo, hi);
            }
            // B fragments: lanes 0-15 col=lane K0-7,K16-23; lanes 16-31 K8-15,K24-31
            v16bf bfr[2];
#pragma unroll
            for (int ni = 0; ni < 2; ++ni) {
                int base = (wn * 32 + ni * 16 + rsel) * BKT + kk + ksel * 8;
                v8bf lo = *(const v8bf*)(&bbuf[base]);
                v8bf hi = *(const v8bf*)(&bbuf[base + 16]);
                bfr[ni] = cat8(lo, hi);
            }
#pragma unroll
            for (int mi = 0; mi < 4; ++mi)
#pragma unroll
                for (int ni = 0; ni < 2; ++ni)
                    acc[mi][ni] = __builtin_amdgcn_wmma_f32_16x16x32_bf16(
                        false, af[mi], false, bfr[ni], (short)0, acc[mi][ni],
                        false, false);
        }

        if (it + 1 < nIter) async_stage_wait();  // next buffer fully landed
        __syncthreads();                         // all waves done with cur buf
    }

    // store: C/D layout -> lanes 0-15: M=v, N=lane; lanes 16-31: M=v+8
#pragma unroll
    for (int mi = 0; mi < 4; ++mi) {
#pragma unroll
        for (int ni = 0; ni < 2; ++ni) {
            int mbase = m0 + wm * 64 + mi * 16 + ksel * 8;
            int n     = n0 + wn * 32 + ni * 16 + rsel;
#pragma unroll
            for (int v = 0; v < 8; ++v) {
                int   m   = mbase + v;
                float val = acc[mi][ni][v];
                if (cmode == 0) {
                    ((__bf16*)C)[bz * strideC + (long)m * N + n] = (__bf16)val;
                } else if (cmode == 1) {
                    long b  = m >> 11;
                    long tk = m & 2047;
                    ((__bf16*)C)[b * ((long)N * 2048) + (long)n * 2048 + tk] =
                        (__bf16)val;
                } else {
                    ((float*)C)[bz * strideC + (long)m * N + n] = val;
                }
            }
        }
    }
}

// ---------------------- softmax over the QUERY axis ------------------------
// s[b][q][k]: normalize over q for each (b,k). In-place bf16. Block (32,8).

__global__ __launch_bounds__(256)
void softmax_qaxis(__bf16* __restrict__ S, int T) {
    const int b = blockIdx.z;
    __bf16* s = S + (long)b * T * T;
    const int k  = blockIdx.x * 32 + threadIdx.x;
    const int ty = threadIdx.y;

    __shared__ float red[8][32];

    float mx = -3.0e38f;
    for (int q = ty; q < T; q += 8)
        mx = fmaxf(mx, (float)s[(long)q * T + k]);
    red[ty][threadIdx.x] = mx;
    __syncthreads();
    if (ty == 0) {
        float m = red[0][threadIdx.x];
#pragma unroll
        for (int i = 1; i < 8; ++i) m = fmaxf(m, red[i][threadIdx.x]);
        red[0][threadIdx.x] = m;
    }
    __syncthreads();
    mx = red[0][threadIdx.x];
    __syncthreads();

    float sum = 0.0f;
    for (int q = ty; q < T; q += 8)
        sum += __expf((float)s[(long)q * T + k] - mx);
    red[ty][threadIdx.x] = sum;
    __syncthreads();
    if (ty == 0) {
        float t = 0.0f;
#pragma unroll
        for (int i = 0; i < 8; ++i) t += red[i][threadIdx.x];
        red[0][threadIdx.x] = t;
    }
    __syncthreads();
    const float inv = 1.0f / red[0][threadIdx.x];

    for (int q = ty; q < T; q += 8) {
        float e = __expf((float)s[(long)q * T + k] - mx);
        s[(long)q * T + k] = (__bf16)(e * inv);
    }
}

// ------------------------------- launcher ----------------------------------

extern "C" void kernel_launch(void* const* d_in, const int* in_sizes, int n_in,
                              void* d_out, int out_size, void* d_ws, size_t ws_size,
                              hipStream_t stream) {
    (void)in_sizes; (void)n_in; (void)out_size; (void)ws_size;

    const int B = 4, T = 2048, D = 1024;
    const int M = B * T;                 // 8192 tokens

    const float* x  = (const float*)d_in[0];
    const float* Wq = (const float*)d_in[1];
    const float* Wk = (const float*)d_in[2];
    const float* Wv = (const float*)d_in[3];
    float* out = (float*)d_out;

    // workspace layout (bytes); total ~107 MB
    char* ws = (char*)d_ws;
    __bf16* xb  = (__bf16*)(ws);                          // 8192x1024      16 MB
    __bf16* wtq = (__bf16*)(ws + 16777216);               // 1024x1024 (T)   2 MB
    __bf16* wtk = wtq + 1048576;
    __bf16* wtv = wtk + 1048576;
    __bf16* qb  = (__bf16*)(ws + 16777216 + 6291456);     // 8192x1024      16 MB
    __bf16* kb  = qb + (long)M * D;                       //                16 MB
    __bf16* vt  = kb + (long)M * D;                       // [4][1024][2048]16 MB
    __bf16* Sb  = vt + (long)M * D;                       // [4][2048][2048]32 MB

    // 1) fp32 -> bf16 conversions (weights transposed: Wt[n][k])
    cvt_f32_bf16<<<dim3((M * D) / 1024), dim3(256), 0, stream>>>(x, xb, M * D);
    cvt_f32_bf16_t<<<dim3((D * D) / 256), dim3(256), 0, stream>>>(Wq, wtq, D, D);
    cvt_f32_bf16_t<<<dim3((D * D) / 256), dim3(256), 0, stream>>>(Wk, wtk, D, D);
    cvt_f32_bf16_t<<<dim3((D * D) / 256), dim3(256), 0, stream>>>(Wv, wtv, D, D);

    // 2) projections: [8192,1024] x [1024,1024]^T
    dim3 gproj(D / BN, M / BM, 1);
    gemm_nt_wmma<<<gproj, dim3(256), 0, stream>>>(xb, wtq, qb, M, D, D, 0, 0, 0, 0);
    gemm_nt_wmma<<<gproj, dim3(256), 0, stream>>>(xb, wtk, kb, M, D, D, 0, 0, 0, 0);
    // V stored transposed per batch: vt[b][d][tk]
    gemm_nt_wmma<<<gproj, dim3(256), 0, stream>>>(xb, wtv, vt, M, D, D, 0, 0, 0, 1);

    // 3) scores S[b] = Q[b] x K[b]^T (naturally NT), bf16 out
    dim3 gscore(T / BN, T / BM, B);
    gemm_nt_wmma<<<gscore, dim3(256), 0, stream>>>(
        qb, kb, Sb, T, T, D,
        (long)T * D, (long)T * D, (long)T * T, 0);

    // 4) softmax over the query axis (axis=1), in place -> P (bf16)
    softmax_qaxis<<<dim3(T / 32, 1, B), dim3(32, 8), 0, stream>>>(Sb, T);

    // 5) O[b] = P[b] x Vt[b]^T, f32 out
    dim3 gout(D / BN, T / BM, B);
    gemm_nt_wmma<<<gout, dim3(256), 0, stream>>>(
        Sb, vt, out, T, D, T,
        (long)T * T, (long)D * T, (long)T * D, 2);
}
